// LPKT_21320217657965
// MI455X (gfx1250) — compile-verified, matrix operand order checked by reading, not run on hardware
//
#include <hip/hip_runtime.h>
#include <hip/hip_bf16.h>
#include <math.h>

// LPKT recurrent knowledge-tracing forward, MI455X (gfx1250, wave32).
// One workgroup (256 thr = 8 waves) per batch row. h[b] (128x64 f32) lives in
// LDS (row stride padded to 68 f32 to kill bank conflicts) for the whole
// 127-step scan. The dominant per-step GEMM h_pre @ W4a ([128,64]x[64,64])
// runs on v_wmma_f32_16x16x32_bf16 with W4a pre-packed in registers;
// LG@W4b + it@W4c folded into a broadcast bias (3x FLOP cut vs x3@W4).
// Fast sigmoid/tanh via v_exp_f32 + v_rcp_f32. Gather rows are ping-pong
// buffered across steps (q_matrix / E_q rows loaded once, reused twice) and
// next-step rows are prefetched (global_prefetch_b8) to hide latency on the
// serial scan chain.

#define Bb 128
#define Ss 128
#define Cc 128
#define Dd 64
#define HS 68   // padded LDS row stride for h (68*4B = 272B, 16B aligned)

typedef __attribute__((ext_vector_type(16))) __bf16 v16bf;
typedef __attribute__((ext_vector_type(8)))  float  v8f;

__device__ __forceinline__ float fast_rcp(float x) {
#if __has_builtin(__builtin_amdgcn_rcpf)
    return __builtin_amdgcn_rcpf(x);
#else
    return 1.0f / x;
#endif
}
__device__ __forceinline__ float sigm(float x) {
    return fast_rcp(1.0f + __expf(-x));          // v_exp_f32 + v_rcp_f32
}
__device__ __forceinline__ float fast_tanh(float x) {
    const float e2 = __expf(2.0f * x);           // tanh = 1 - 2/(e^{2x}+1)
    return 1.0f - 2.0f * fast_rcp(e2 + 1.0f);
}

__global__ __launch_bounds__(256) void lpkt_scan_kernel(
    const int* __restrict__ qseq, const int* __restrict__ cseq,
    const int* __restrict__ itseq, const int* __restrict__ atseq,
    const float* __restrict__ E_q, const float* __restrict__ E_c,
    const float* __restrict__ E_it, const float* __restrict__ E_at,
    const float* __restrict__ q_matrix, const float* __restrict__ h0,
    const float* __restrict__ W1, const float* __restrict__ b1,
    const float* __restrict__ W2, const float* __restrict__ b2,
    const float* __restrict__ W3, const float* __restrict__ b3,
    const float* __restrict__ W4, const float* __restrict__ b4,
    const float* __restrict__ W5, const float* __restrict__ b5,
    float* __restrict__ pred)
{
    __shared__ float h[Cc * HS];        // knowledge state, padded rows, ~34KB
    __shared__ float x4[4 * Dd];        // [learn_prev | it | learn_t | h_tilde_prev]
    __shared__ float eac[3 * Dd];       // [ - | at | c ] (e part lives in enxbuf)
    __shared__ float LG[Dd];
    __shared__ float biasbc[Dd];        // LG@W4b + it@W4c + b4 (broadcast over C)
    __shared__ float htil[Dd];
    __shared__ float zv[Dd];
    __shared__ float tg[Dd];            // W2 gate pre-activation (split phase)
    __shared__ float ts[Dd];            // W3 gate pre-activation (split phase)
    __shared__ float qrowbuf[2][Cc];    // ping-pong q_matrix rows (t / t+1)
    __shared__ float enxbuf[2][Dd];     // ping-pong E_q rows (t / t+1)

    const int b    = blockIdx.x;
    const int tid  = threadIdx.x;
    const int wave = tid >> 5;
    const int lane = tid & 31;
    const int m0   = wave * 16;         // this wave owns h rows [m0, m0+16)

    // ---- Pre-pack W4a (rows 0..63 of W4) as bf16 WMMA B-operands: 4 N-tiles x 2 K-slices.
    // B layout mirrors ISA 16-bit A layout with N in place of M:
    //   lanes 0-15 : N = n0+lane, elems 0..7 -> K=kb..kb+7, elems 8..15 -> K=kb+16..+23, kb = kh
    //   lanes 16-31: same N, kb = kh + 8
    v16bf Bop[4][2];
    {
        const int nl   = lane & 15;
        const int kofs = (lane < 16) ? 0 : 8;
#pragma unroll
        for (int nt = 0; nt < 4; ++nt) {
            const int n = nt * 16 + nl;
#pragma unroll
            for (int kh2 = 0; kh2 < 2; ++kh2) {
                const int kb = kh2 * 32 + kofs;
#pragma unroll
                for (int i = 0; i < 8; ++i) {
                    Bop[nt][kh2][i]     = (__bf16)W4[(kb + i)      * Dd + n];
                    Bop[nt][kh2][8 + i] = (__bf16)W4[(kb + 16 + i) * Dd + n];
                }
            }
        }
    }

    // ---- Init: h = h0, learning_prev = 0, pred[b,0] = 0, row/e buffers for t=0,
    //      h_tilde0 = q_e[0] @ h0
    const int q0 = qseq[b * Ss];
    for (int i = tid; i < Cc * Dd; i += 256) h[(i >> 6) * HS + (i & 63)] = h0[i];
    if (tid < Dd) x4[tid] = 0.0f;                       // learning_prev at t=0
    if (tid < Cc) qrowbuf[0][tid] = q_matrix[(size_t)q0 * Cc + tid];
    if (tid >= 128 && tid < 192) enxbuf[0][tid - 128] = E_q[(size_t)q0 * Dd + (tid - 128)];
    if (tid == 0) pred[b * Ss] = 0.0f;
    __syncthreads();
    if (tid < Dd) {
        float s0 = 0.f, s1 = 0.f;
        for (int c2 = 0; c2 < Cc; c2 += 2) {
            s0 += qrowbuf[0][c2]     * h[c2 * HS + tid];
            s1 += qrowbuf[0][c2 + 1] * h[(c2 + 1) * HS + tid];
        }
        htil[tid] = s0 + s1;
    }
    __syncthreads();

    // ---- 127-step scan
    for (int t = 0; t < Ss - 1; ++t) {
        const int cur = t & 1, nxt = cur ^ 1;
        const int qt1 = qseq[b * Ss + t + 1];

        // Phase 0: fresh gathers (only t+1 rows are new) + carry copies + prefetch
        if (tid < 64) {
            x4[192 + tid]    = htil[tid];                           // h_tilde_prev
            enxbuf[nxt][tid] = E_q[(size_t)qt1 * Dd + tid];         // e embed at t+1
        } else if (tid < 128) {
            eac[tid] = E_at[(size_t)atseq[b * Ss + t] * Dd + (tid - 64)];
        } else if (tid < 192) {
            eac[tid] = E_c[(size_t)cseq[b * Ss + t] * Dd + (tid - 128)];
        } else {
            x4[64 + (tid - 192)] = E_it[(size_t)itseq[b * Ss + t] * Dd + (tid - 192)];
        }
        if (tid < 128) {
            qrowbuf[nxt][tid] = q_matrix[(size_t)qt1 * Cc + tid];
        } else {
            // Prefetch the rows step t+1's phase 0 will gather (hide HBM/L2 latency
            // behind this step's compute on the serial chain).
            const int lid = tid - 128;
            const int t2  = (t + 2 < Ss) ? (t + 2) : (Ss - 1);
            if (lid < 2) {
                const int q2 = qseq[b * Ss + t2];
                __builtin_prefetch((const char*)(E_q + (size_t)q2 * Dd) + lid * 128, 0, 3);
            } else if (lid < 6) {
                const int q2 = qseq[b * Ss + t2];
                __builtin_prefetch((const char*)(q_matrix + (size_t)q2 * Cc) + (lid - 2) * 128, 0, 3);
            } else if (lid < 8) {
                const int at1 = atseq[b * Ss + t + 1];
                __builtin_prefetch((const char*)(E_at + (size_t)at1 * Dd) + (lid - 6) * 128, 0, 3);
            } else if (lid < 10) {
                const int it1 = itseq[b * Ss + t + 1];
                __builtin_prefetch((const char*)(E_it + (size_t)it1 * Dd) + (lid - 8) * 128, 0, 3);
            }
        }
        __syncthreads();

        // Phase 1: learning_t = [e|at|c] @ W1 + b1  -> x4[128..191]
        if (tid < Dd) {
            const float* ecur = enxbuf[cur];
            float a0 = 0.f, a1 = 0.f, a2 = 0.f, a3 = 0.f;
            for (int k = 0; k < Dd; k += 4) {               // e block (rows 0..63)
                a0 += ecur[k]     * W1[k * Dd + tid];
                a1 += ecur[k + 1] * W1[(k + 1) * Dd + tid];
                a2 += ecur[k + 2] * W1[(k + 2) * Dd + tid];
                a3 += ecur[k + 3] * W1[(k + 3) * Dd + tid];
            }
            for (int k = Dd; k < 3 * Dd; k += 4) {          // at | c blocks
                a0 += eac[k]     * W1[k * Dd + tid];
                a1 += eac[k + 1] * W1[(k + 1) * Dd + tid];
                a2 += eac[k + 2] * W1[(k + 2) * Dd + tid];
                a3 += eac[k + 3] * W1[(k + 3) * Dd + tid];
            }
            x4[128 + tid] = a0 + a1 + a2 + a3 + b1[tid];
        }
        __syncthreads();

        // Phase 2 (split): tg = x4@W2+b2 on threads 0-63, ts = x4@W3+b3 on 64-127
        if (tid < 64) {
            const int d = tid;
            float a0 = 0.f, a1 = 0.f, a2 = 0.f, a3 = 0.f;
            for (int k = 0; k < 4 * Dd; k += 4) {
                a0 += x4[k]     * W2[k * Dd + d];
                a1 += x4[k + 1] * W2[(k + 1) * Dd + d];
                a2 += x4[k + 2] * W2[(k + 2) * Dd + d];
                a3 += x4[k + 3] * W2[(k + 3) * Dd + d];
            }
            tg[d] = a0 + a1 + a2 + a3 + b2[d];
        } else if (tid < 128) {
            const int d = tid - 64;
            float a0 = 0.f, a1 = 0.f, a2 = 0.f, a3 = 0.f;
            for (int k = 0; k < 4 * Dd; k += 4) {
                a0 += x4[k]     * W3[k * Dd + d];
                a1 += x4[k + 1] * W3[(k + 1) * Dd + d];
                a2 += x4[k + 2] * W3[(k + 2) * Dd + d];
                a3 += x4[k + 3] * W3[(k + 3) * Dd + d];
            }
            ts[d] = a0 + a1 + a2 + a3 + b3[d];
        }
        __syncthreads();

        // Phase 3: LG = sigmoid(ts) * (tanh(tg)+1)/2
        if (tid < Dd) {
            LG[tid] = sigm(ts[tid]) * (fast_tanh(tg[tid]) + 1.0f) * 0.5f;
        }
        __syncthreads();

        // Phase 4: broadcast bias = LG@W4b + it@W4c + b4
        if (tid < Dd) {
            float a0 = 0.f, a1 = 0.f;
            for (int k = 0; k < Dd; ++k) {
                a0 += LG[k]      * W4[(64 + k)  * Dd + tid];
                a1 += x4[64 + k] * W4[(128 + k) * Dd + tid];
            }
            biasbc[tid] = a0 + a1 + b4[tid];
        }
        __syncthreads();

        // Phase 5: G = h_bf16 @ W4a via WMMA, fused gate + state update.
        {
            // A operands (16x32 bf16 each) from this wave's 16 h rows.
            const int m = m0 + (lane & 15);
            const float* row = &h[m * HS];
            const int lo = (lane < 16) ? 0 : 8;
            v16bf A0, A1;
#pragma unroll
            for (int i = 0; i < 8; ++i) {
                A0[i]     = (__bf16)row[lo + i];
                A0[8 + i] = (__bf16)row[16 + lo + i];
                A1[i]     = (__bf16)row[32 + lo + i];
                A1[8 + i] = (__bf16)row[48 + lo + i];
            }
            const int mb = m0 + ((lane >> 4) << 3);   // f32 C/D layout: M = mb + r
            const float* qr = qrowbuf[cur];
#pragma unroll
            for (int nt = 0; nt < 4; ++nt) {
                v8f acc = {};
                acc = __builtin_amdgcn_wmma_f32_16x16x32_bf16(
                          false, A0, false, Bop[nt][0], (short)0, acc, false, false);
                acc = __builtin_amdgcn_wmma_f32_16x16x32_bf16(
                          false, A1, false, Bop[nt][1], (short)0, acc, false, false);
                const int n   = nt * 16 + (lane & 15);
                const float lgn = LG[n], bb = biasbc[n];
#pragma unroll
                for (int r = 0; r < 8; ++r) {
                    const int mm = mb + r;
                    const float gf   = sigm(acc[r] + bb);
                    const float hold = h[mm * HS + n];
                    h[mm * HS + n] = qr[mm] * lgn + gf * hold;
                }
            }
        }
        __syncthreads();

        // Phase 6: h_tilde = q_e_next @ h ; shift learning carry
        if (tid < Dd) {
            const float* qn = qrowbuf[nxt];
            float s0 = 0.f, s1 = 0.f;
            for (int c2 = 0; c2 < Cc; c2 += 2) {
                s0 += qn[c2]     * h[c2 * HS + tid];
                s1 += qn[c2 + 1] * h[(c2 + 1) * HS + tid];
            }
            htil[tid] = s0 + s1;
        } else if (tid < 128) {
            x4[tid - 64] = x4[tid + 64];   // learning_prev <- learning_t
        }
        __syncthreads();

        // Phase 7: prediction head z = sigmoid([e_next | h_tilde] @ W5 + b5)
        if (tid < Dd) {
            const float* en = enxbuf[nxt];
            float a0 = 0.f, a1 = 0.f;
            for (int k = 0; k < Dd; ++k) {
                a0 += en[k]   * W5[k * Dd + tid];
                a1 += htil[k] * W5[(64 + k) * Dd + tid];
            }
            zv[tid] = sigm(a0 + a1 + b5[tid]);
        }
        __syncthreads();

        // Phase 8: mean over D -> pred[b, t+1]  (wave-level tree reduction)
        if (tid < 32) {
            float s = zv[tid] + zv[tid + 32];
            for (int o = 16; o > 0; o >>= 1) s += __shfl_down(s, o, 32);
            if (tid == 0) pred[b * Ss + t + 1] = s * (1.0f / Dd);
        }
        __syncthreads();
    }
}

extern "C" void kernel_launch(void* const* d_in, const int* in_sizes, int n_in,
                              void* d_out, int out_size, void* d_ws, size_t ws_size,
                              hipStream_t stream) {
    (void)in_sizes; (void)n_in; (void)d_ws; (void)ws_size; (void)out_size;
    const int*   qseq  = (const int*)  d_in[0];
    const int*   cseq  = (const int*)  d_in[1];
    const int*   itseq = (const int*)  d_in[2];
    const int*   atseq = (const int*)  d_in[3];
    const float* E_q   = (const float*)d_in[4];
    const float* E_c   = (const float*)d_in[5];
    const float* E_it  = (const float*)d_in[6];
    const float* E_at  = (const float*)d_in[7];
    const float* qmat  = (const float*)d_in[8];
    const float* h0    = (const float*)d_in[9];
    const float* W1    = (const float*)d_in[10];
    const float* b1    = (const float*)d_in[11];
    const float* W2    = (const float*)d_in[12];
    const float* b2    = (const float*)d_in[13];
    const float* W3    = (const float*)d_in[14];
    const float* b3    = (const float*)d_in[15];
    const float* W4    = (const float*)d_in[16];
    const float* b4    = (const float*)d_in[17];
    const float* W5    = (const float*)d_in[18];
    const float* b5    = (const float*)d_in[19];
    float* pred = (float*)d_out;

    lpkt_scan_kernel<<<dim3(Bb), dim3(256), 0, stream>>>(
        qseq, cseq, itseq, atseq, E_q, E_c, E_it, E_at, qmat, h0,
        W1, b1, W2, b2, W3, b3, W4, b4, W5, b5, pred);
}
